// SRNN_62766652064123
// MI455X (gfx1250) — compile-verified, matrix operand order checked by reading, not run on hardware
//
#include <hip/hip_runtime.h>
#include <hip/hip_bf16.h>

// ---------------------------------------------------------------------------
// SRNN forward for MI455X (gfx1250, wave32, WMMA).
// All GEMMs use v_wmma_f32_16x16x32_bf16 (bf16 in, f32 accumulate).
// Fused LSTM: 32-row M-tile (2 A fragments per B fragment -> 2 WMMA per B
// load, half the L2 weight traffic), gates staged in a 64KB LDS tile reused
// across two pointwise phases.
// ---------------------------------------------------------------------------

typedef __bf16 bf16;
typedef __attribute__((ext_vector_type(16))) __bf16 v16bf;
typedef __attribute__((ext_vector_type(8)))  float   v8f;

#define NW      192                 // N nodes
#define TSTEPS  12
#define NN      (NW * NW)           // 36864 edges
#define ERNN    256
#define NRNN    256
#define EEMB    128
#define NEMB    256
#define DECD    256
#define NOUT    5

// ------------------------------ helpers -----------------------------------

__device__ __forceinline__ bf16 f2bf(float x) {
    union { float f; unsigned u; } a; a.f = x;
    unsigned r = a.u + 0x7FFFu + ((a.u >> 16) & 1u);   // round-to-nearest-even
    union { unsigned short s; bf16 b; } c; c.s = (unsigned short)(r >> 16);
    return c.b;
}
__device__ __forceinline__ float bf2f(bf16 b) {
    union { unsigned short s; bf16 b; } a; a.b = b;
    union { unsigned u; float f; } c; c.u = ((unsigned)a.s) << 16;
    return c.f;
}
__device__ __forceinline__ float sigf(float x) { return 1.0f / (1.0f + __expf(-x)); }

// ISA 7.12.2: 16-bit A 16x32 fragment. lane<16: row=lane, K=base+{0..7,16..23};
// lane>=16: row=lane-16, K=base+{8..15,24..31}. Two 16B loads per lane.
// Same layout for B (32x16) with lane -> column of W[N x K] row-major (= B^T row).
__device__ __forceinline__ v16bf load_frag(const bf16* __restrict__ base, int pitch,
                                           int r0, int kb, int lane) {
    int r = r0 + (lane & 15);
    int k = kb + ((lane >> 4) << 3);
    const bf16* p = base + (size_t)r * (size_t)pitch + k;
    union { uint4 u[2]; v16bf v; } f;
    f.u[0] = *reinterpret_cast<const uint4*>(p);
    f.u[1] = *reinterpret_cast<const uint4*>(p + 16);
    return f.v;
}

// ------------------------------ small kernels ------------------------------

__global__ void f32_to_bf16_kernel(const float* __restrict__ in, bf16* __restrict__ out, int n) {
    int i = blockIdx.x * blockDim.x + threadIdx.x;
    if (i < n) out[i] = f2bf(in[i]);
}

__global__ void gather_rows_f32_kernel(const float* __restrict__ src, float* __restrict__ dst,
                                       int rows, int cols, int srcPitch) {
    int i = blockIdx.x * blockDim.x + threadIdx.x;
    if (i >= rows * cols) return;
    int r = i / cols, c = i - r * cols;
    dst[i] = src[(size_t)r * srcPitch + c];
}

// e = relu(x @ w^T + b), x is [M x 2] with row pitch ldx. Output bf16, pitch ldo.
__global__ void embed2_kernel(const float* __restrict__ x, int ldx,
                              const float* __restrict__ w, const float* __restrict__ b,
                              bf16* __restrict__ out, int ldo, int M, int NO) {
    int i = blockIdx.x * blockDim.x + threadIdx.x;
    if (i >= M * NO) return;
    int m = i / NO, j = i - m * NO;
    float v = x[(size_t)m * ldx] * w[2 * j] + x[(size_t)m * ldx + 1] * w[2 * j + 1] + b[j];
    out[(size_t)m * ldo + j] = f2bf(fmaxf(v, 0.0f));
}

// out[m,o] = dec[m,:] . wo[o,:] + bo[o]   (tiny: 192x5)
__global__ void out5_kernel(const bf16* __restrict__ dec, const float* __restrict__ wo,
                            const float* __restrict__ bo, float* __restrict__ out) {
    int i = blockIdx.x * blockDim.x + threadIdx.x;
    if (i >= NW * NOUT) return;
    int m = i / NOUT, o = i - m * NOUT;
    const bf16* d = dec + (size_t)m * DECD;
    const float* w = wo + (size_t)o * DECD;
    float s = bo[o];
    for (int k = 0; k < DECD; ++k) s += bf2f(d[k]) * w[k];
    out[i] = s;
}

// ------------------------------ WMMA GEMM ----------------------------------
// D[M x N] = act(A[M x K] @ W[N x K]^T + bias)
// grid.x = M/(16*MT), grid.y = N/(8*TPW*16). 8 waves/WG; each wave owns
// TPW column tiles x MT row tiles; every B fragment feeds MT WMMAs.
template <int TPW, int MT>
__global__ __launch_bounds__(256) void gemm_bf16_kernel(
        const bf16* __restrict__ A, int lda, int K,
        const bf16* __restrict__ W, const float* __restrict__ bias,
        float* __restrict__ outF, int ldoF,
        bf16*  __restrict__ outB, int ldoB, int do_relu) {
    const int lane = threadIdx.x & 31;
    const int wid  = threadIdx.x >> 5;
    const int row0 = blockIdx.x * (16 * MT);
    const int n0   = (blockIdx.y * 8 + wid) * (TPW * 16);

    v8f acc[MT][TPW];
#pragma unroll
    for (int m = 0; m < MT; ++m)
#pragma unroll
        for (int t = 0; t < TPW; ++t)
            acc[m][t] = (v8f){0.f,0.f,0.f,0.f,0.f,0.f,0.f,0.f};

    for (int kb = 0; kb < K; kb += 32) {
        v16bf a[MT];
#pragma unroll
        for (int m = 0; m < MT; ++m) a[m] = load_frag(A, lda, row0 + m * 16, kb, lane);
#pragma unroll
        for (int t = 0; t < TPW; ++t) {
            v16bf b = load_frag(W, K, n0 + t * 16, kb, lane);
#pragma unroll
            for (int m = 0; m < MT; ++m)
                acc[m][t] = __builtin_amdgcn_wmma_f32_16x16x32_bf16(
                    false, a[m], false, b, (short)0, acc[m][t], false, false);
        }
    }
    const int rlo = (lane >> 4) << 3;   // C/D: vgpr r -> row r (+8 for lanes 16..31)
    const int cl  = lane & 15;
#pragma unroll
    for (int m = 0; m < MT; ++m)
#pragma unroll
        for (int t = 0; t < TPW; ++t) {
            int col = n0 + t * 16 + cl;
            float bv = bias ? bias[col] : 0.0f;
#pragma unroll
            for (int r = 0; r < 8; ++r) {
                int row = row0 + m * 16 + rlo + r;
                float v = acc[m][t][r] + bv;
                if (do_relu) v = fmaxf(v, 0.0f);
                if (outF) outF[(size_t)row * ldoF + col] = v;
                if (outB) outB[(size_t)row * ldoB + col] = f2bf(v);
            }
        }
}

// --------------------- fused LSTM: gates GEMM + pointwise ------------------
// gates[32 x 1024] = A1 @ W1^T + A2 @ W2^T  (bias added in pointwise)
// then c' = sig(f)*c + sig(i)*tanh(g); h = sig(o)*tanh(c').
// grid.x = M/32; 8 waves each own 128 gate columns; 2 row tiles share each
// B fragment. Pointwise runs in two 16-row phases reusing one 64KB LDS tile.
__global__ __launch_bounds__(256) void lstm_fused_kernel(
        const bf16* __restrict__ A1, int lda1, int K1, const bf16* __restrict__ W1,
        const bf16* __restrict__ A2, int lda2, int K2, const bf16* __restrict__ W2,
        const float* __restrict__ bih, const float* __restrict__ bhh,
        const float* __restrict__ c_in, int ldci,
        float* __restrict__ c_out, int ldco,
        float* __restrict__ h_out, int ldho,
        bf16*  __restrict__ hb_out, int ldhb) {
    __shared__ float gbuf[16 * 1024];   // 64 KB (of 320 KB/WGP), reused twice

    const int lane = threadIdx.x & 31;
    const int wid  = threadIdx.x >> 5;
    const int row0 = blockIdx.x * 32;
    const int n0   = wid * 128;

    v8f acc[2][8];
#pragma unroll
    for (int m = 0; m < 2; ++m)
#pragma unroll
        for (int t = 0; t < 8; ++t)
            acc[m][t] = (v8f){0.f,0.f,0.f,0.f,0.f,0.f,0.f,0.f};

    for (int kb = 0; kb < K1; kb += 32) {
        v16bf a0 = load_frag(A1, lda1, row0,      kb, lane);
        v16bf a1 = load_frag(A1, lda1, row0 + 16, kb, lane);
#pragma unroll
        for (int t = 0; t < 8; ++t) {
            v16bf b = load_frag(W1, K1, n0 + t * 16, kb, lane);
            acc[0][t] = __builtin_amdgcn_wmma_f32_16x16x32_bf16(
                false, a0, false, b, (short)0, acc[0][t], false, false);
            acc[1][t] = __builtin_amdgcn_wmma_f32_16x16x32_bf16(
                false, a1, false, b, (short)0, acc[1][t], false, false);
        }
    }
    for (int kb = 0; kb < K2; kb += 32) {
        v16bf a0 = load_frag(A2, lda2, row0,      kb, lane);
        v16bf a1 = load_frag(A2, lda2, row0 + 16, kb, lane);
#pragma unroll
        for (int t = 0; t < 8; ++t) {
            v16bf b = load_frag(W2, K2, n0 + t * 16, kb, lane);
            acc[0][t] = __builtin_amdgcn_wmma_f32_16x16x32_bf16(
                false, a0, false, b, (short)0, acc[0][t], false, false);
            acc[1][t] = __builtin_amdgcn_wmma_f32_16x16x32_bf16(
                false, a1, false, b, (short)0, acc[1][t], false, false);
        }
    }

    const int rlo = (lane >> 4) << 3;
    const int cl  = lane & 15;

#pragma unroll
    for (int rt = 0; rt < 2; ++rt) {
        // spill this row tile's gates to LDS
#pragma unroll
        for (int t = 0; t < 8; ++t) {
            int col = n0 + t * 16 + cl;
#pragma unroll
            for (int r = 0; r < 8; ++r) gbuf[(rlo + r) * 1024 + col] = acc[rt][t][r];
        }
        __syncthreads();

        // 16 rows x 256 hidden = 4096 items over 256 threads
#pragma unroll 4
        for (int it = 0; it < 16; ++it) {
            int idx = threadIdx.x + it * 256;
            int row = idx >> 8;
            int j   = idx & 255;
            float gi = gbuf[row * 1024 +   0 + j] + bih[  0 + j] + bhh[  0 + j];
            float gf = gbuf[row * 1024 + 256 + j] + bih[256 + j] + bhh[256 + j];
            float gg = gbuf[row * 1024 + 512 + j] + bih[512 + j] + bhh[512 + j];
            float go = gbuf[row * 1024 + 768 + j] + bih[768 + j] + bhh[768 + j];
            int grow = row0 + rt * 16 + row;
            float co = c_in[(size_t)grow * ldci + j];
            float cn = sigf(gf) * co + sigf(gi) * tanhf(gg);
            float h  = sigf(go) * tanhf(cn);
            c_out[(size_t)grow * ldco + j] = cn;
            h_out[(size_t)grow * ldho + j] = h;
            if (hb_out) hb_out[(size_t)grow * ldhb + j] = f2bf(h);
        }
        __syncthreads();   // protect gbuf before next phase overwrites it
    }
}

// ------------------------------ attention ----------------------------------
// per source node i: logits_j = (N-1) * <hs[i,j,:], q[i,:]>, softmax over j!=i,
// h_sp[i,:] = sum_j a_j * hs[i,j,:]  -> written as bf16 into xin[:, 512:768].
__global__ __launch_bounds__(256) void attn_kernel(const float* __restrict__ hs,
                                                   const float* __restrict__ q,
                                                   bf16* __restrict__ xin) {
    __shared__ float qv[256];
    __shared__ float lg[NW];
    __shared__ float red[256];
    const int i   = blockIdx.x;
    const int tid = threadIdx.x;
    qv[tid] = q[i * 256 + tid];
    __syncthreads();

    const int wid = tid >> 5, lane = tid & 31;
    for (int j = wid; j < NW; j += 8) {
        const float* row = hs + ((size_t)i * NW + j) * 256;
        float p = 0.f;
        for (int k = lane; k < 256; k += 32) p += row[k] * qv[k];
        for (int off = 16; off; off >>= 1) p += __shfl_xor(p, off, 32);
        if (lane == 0) lg[j] = p * (float)(NW - 1);
    }
    __syncthreads();

    float mv = -3.4e38f;
    if (tid < NW && tid != i) mv = lg[tid];
    red[tid] = mv;
    __syncthreads();
    for (int s = 128; s; s >>= 1) { if (tid < s) red[tid] = fmaxf(red[tid], red[tid + s]); __syncthreads(); }
    float mx = red[0];
    __syncthreads();

    float ev = 0.f;
    if (tid < NW && tid != i) ev = __expf(lg[tid] - mx);
    red[tid] = ev;
    __syncthreads();
    for (int s = 128; s; s >>= 1) { if (tid < s) red[tid] += red[tid + s]; __syncthreads(); }
    float inv = 1.0f / red[0];
    __syncthreads();
    if (tid < NW) lg[tid] = ev * inv;
    __syncthreads();

    float acc = 0.f;
    for (int j = 0; j < NW; ++j)
        acc += lg[j] * hs[((size_t)i * NW + j) * 256 + tid];
    xin[(size_t)i * 768 + 512 + tid] = f2bf(acc);
}

// ------------------------------ host launch --------------------------------

extern "C" void kernel_launch(void* const* d_in, const int* in_sizes, int n_in,
                              void* d_out, int out_size, void* d_ws, size_t ws_size,
                              hipStream_t stream) {
    (void)in_sizes; (void)n_in; (void)out_size; (void)ws_size;

    // ---- inputs (setup_inputs order; params pytree flattened by sorted keys)
    const float* nodes  = (const float*)d_in[0];   // [T,N,2]
    const float* edges  = (const float*)d_in[1];   // [T,N,N,2]
    const float* hn_in  = (const float*)d_in[2];
    const float* cn_in  = (const float*)d_in[3];
    const float* he_in  = (const float*)d_in[4];
    const float* ce_in  = (const float*)d_in[5];
    const float* attn_w = (const float*)d_in[6];   // [256,256]
    // node: bd, be, bhh, bih, bo, wd, we, whh, wih, wo
    const float* n_bd  = (const float*)d_in[7];
    const float* n_be  = (const float*)d_in[8];
    const float* n_bhh = (const float*)d_in[9];
    const float* n_bih = (const float*)d_in[10];
    const float* n_bo  = (const float*)d_in[11];
    const float* n_wd  = (const float*)d_in[12];
    const float* n_we  = (const float*)d_in[13];
    const float* n_whh = (const float*)d_in[14];
    const float* n_wih = (const float*)d_in[15];
    const float* n_wo  = (const float*)d_in[16];
    // spatial: b1, b2, bhh, bih, w1, w2, whh, wih
    const float* s_b1  = (const float*)d_in[17];
    const float* s_b2  = (const float*)d_in[18];
    const float* s_bhh = (const float*)d_in[19];
    const float* s_bih = (const float*)d_in[20];
    const float* s_w1  = (const float*)d_in[21];
    const float* s_w2  = (const float*)d_in[22];
    const float* s_whh = (const float*)d_in[23];
    const float* s_wih = (const float*)d_in[24];
    // temporal: same order
    const float* t_b1  = (const float*)d_in[25];
    const float* t_b2  = (const float*)d_in[26];
    const float* t_bhh = (const float*)d_in[27];
    const float* t_bih = (const float*)d_in[28];
    const float* t_w1  = (const float*)d_in[29];
    const float* t_w2  = (const float*)d_in[30];
    const float* t_whh = (const float*)d_in[31];
    const float* t_wih = (const float*)d_in[32];

    // ---- output regions (states live in-place in d_out)
    float* out_outs = (float*)d_out;                    // [T,N,5]
    float* out_hn   = out_outs + (size_t)TSTEPS * NW * NOUT;
    float* out_he   = out_hn + (size_t)NW * NRNN;
    float* out_cn   = out_he + (size_t)NN * ERNN;
    float* out_ce   = out_cn + (size_t)NW * NRNN;

    // ---- workspace layout
    char* ws = (char*)d_ws;
    size_t off = 0;
    auto abf = [&](size_t e) { size_t o = off; off += ((e * 2 + 255) / 256) * 256; return o; };
    auto af32 = [&](size_t e) { size_t o = off; off += ((e * 4 + 255) / 256) * 256; return o; };
    bf16* w2s_b  = (bf16*)(ws + abf(128 * 128));
    bf16* wihs_b = (bf16*)(ws + abf(1024 * 128));
    bf16* whhs_b = (bf16*)(ws + abf(1024 * 256));
    bf16* w2t_b  = (bf16*)(ws + abf(128 * 128));
    bf16* wiht_b = (bf16*)(ws + abf(1024 * 128));
    bf16* whht_b = (bf16*)(ws + abf(1024 * 256));
    bf16* wihn_b = (bf16*)(ws + abf(1024 * 768));
    bf16* whhn_b = (bf16*)(ws + abf(1024 * 256));
    bf16* wd_b   = (bf16*)(ws + abf(256 * 256));
    bf16* attn_b = (bf16*)(ws + abf(256 * 256));
    bf16* h_bf   = (bf16*)(ws + abf((size_t)NN * ERNN));   // snapshot of h_e
    bf16* hn_bf  = (bf16*)(ws + abf((size_t)NW * NRNN));   // snapshot of h_n
    bf16* e1     = (bf16*)(ws + abf((size_t)NN * EEMB));
    bf16* e2     = (bf16*)(ws + abf((size_t)NN * EEMB));
    bf16* e1t    = (bf16*)(ws + abf((size_t)NW * EEMB));
    bf16* e2t    = (bf16*)(ws + abf((size_t)NW * EEMB));
    bf16* xin    = (bf16*)(ws + abf((size_t)NW * 768));
    bf16* hn2_bf = (bf16*)(ws + abf((size_t)NW * NRNN));
    bf16* dec_bf = (bf16*)(ws + abf((size_t)NW * DECD));
    float* qbuf  = (float*)(ws + af32((size_t)NW * ERNN));
    float* cdiag = (float*)(ws + af32((size_t)NW * ERNN));

    auto cv = [&](const float* src, bf16* dst, int n) {
        f32_to_bf16_kernel<<<(n + 255) / 256, 256, 0, stream>>>(src, dst, n);
    };

    // ---- init states in d_out
    hipMemcpyAsync(out_hn, hn_in, (size_t)NW * NRNN * 4, hipMemcpyDeviceToDevice, stream);
    hipMemcpyAsync(out_cn, cn_in, (size_t)NW * NRNN * 4, hipMemcpyDeviceToDevice, stream);
    hipMemcpyAsync(out_he, he_in, (size_t)NN * ERNN * 4, hipMemcpyDeviceToDevice, stream);
    hipMemcpyAsync(out_ce, ce_in, (size_t)NN * ERNN * 4, hipMemcpyDeviceToDevice, stream);

    // ---- one-time weight conversion to bf16 (L2-resident afterwards)
    cv(s_w2, w2s_b, 128 * 128);   cv(s_wih, wihs_b, 1024 * 128);  cv(s_whh, whhs_b, 1024 * 256);
    cv(t_w2, w2t_b, 128 * 128);   cv(t_wih, wiht_b, 1024 * 128);  cv(t_whh, whht_b, 1024 * 256);
    cv(n_wih, wihn_b, 1024 * 768); cv(n_whh, whhn_b, 1024 * 256);
    cv(n_wd, wd_b, 256 * 256);    cv(attn_w, attn_b, 256 * 256);

    const int DIAG_PITCH = (NW + 1) * ERNN;   // row pitch of diagonal entries

    for (int t = 0; t < TSTEPS; ++t) {
        const float* edg = edges + (size_t)t * NN * 2;
        const float* pos = nodes + (size_t)t * NW * 2;
        float* outs_t = out_outs + (size_t)t * NW * NOUT;

        // --- step-start snapshots (old h as bf16; old diagonal c as f32)
        cv(out_he, h_bf, NN * ERNN);
        cv(out_hn, hn_bf, NW * NRNN);
        gather_rows_f32_kernel<<<(NW * ERNN + 255) / 256, 256, 0, stream>>>(
            out_ce, cdiag, NW, ERNN, DIAG_PITCH);

        // --- spatial edges (all N^2): embed -> e2 -> fused LSTM (in-place he/ce)
        embed2_kernel<<<((size_t)NN * EEMB + 255) / 256, 256, 0, stream>>>(
            edg, 2, s_w1, s_b1, e1, EEMB, NN, EEMB);
        gemm_bf16_kernel<1, 2><<<dim3(NN / 32, 1), 256, 0, stream>>>(
            e1, EEMB, EEMB, w2s_b, s_b2, nullptr, 0, e2, EEMB, 1);
        lstm_fused_kernel<<<NN / 32, 256, 0, stream>>>(
            e2, EEMB, EEMB, wihs_b, h_bf, ERNN, ERNN, whhs_b,
            s_bih, s_bhh, out_ce, ERNN, out_ce, ERNN, out_he, ERNN, nullptr, 0);

        // --- attention over NEW spatial h (uses OLD h_n): q = h_n @ Wn^T
        gemm_bf16_kernel<2, 1><<<dim3(NW / 16, 1), 256, 0, stream>>>(
            hn_bf, NRNN, NRNN, attn_b, nullptr, qbuf, ERNN, nullptr, 0, 0);
        attn_kernel<<<NW, 256, 0, stream>>>(out_he, qbuf, xin);

        // --- temporal edges (diagonal, via strided pointers)
        embed2_kernel<<<(NW * EEMB + 255) / 256, 256, 0, stream>>>(
            edg, (NW + 1) * 2, t_w1, t_b1, e1t, EEMB, NW, EEMB);
        gemm_bf16_kernel<1, 1><<<dim3(NW / 16, 1), 256, 0, stream>>>(
            e1t, EEMB, EEMB, w2t_b, t_b2, nullptr, 0, e2t, EEMB, 1);
        lstm_fused_kernel<<<NW / 32, 256, 0, stream>>>(
            e2t, EEMB, EEMB, wiht_b, h_bf, DIAG_PITCH, ERNN, whht_b,
            t_bih, t_bhh, cdiag, ERNN,
            out_ce, DIAG_PITCH,          // scatter c' onto diag of c_e
            out_he, DIAG_PITCH,          // scatter h' onto diag of h_e
            xin + 256, 768);             // ht (bf16) -> xin[:,256:512]

        // --- node RNN: enc -> xin[:,0:256]; fused LSTM; decoder; output head
        embed2_kernel<<<(NW * NEMB + 255) / 256, 256, 0, stream>>>(
            pos, 2, n_we, n_be, xin, 768, NW, NEMB);
        lstm_fused_kernel<<<NW / 32, 256, 0, stream>>>(
            xin, 768, 768, wihn_b, hn_bf, NRNN, NRNN, whhn_b,
            n_bih, n_bhh, out_cn, NRNN, out_cn, NRNN, out_hn, NRNN,
            hn2_bf, NRNN);
        gemm_bf16_kernel<2, 1><<<dim3(NW / 16, 1), 256, 0, stream>>>(
            hn2_bf, NRNN, NRNN, wd_b, n_bd, nullptr, 0, dec_bf, DECD, 1);
        out5_kernel<<<(NW * NOUT + 255) / 256, 256, 0, stream>>>(
            dec_bf, n_wo, n_bo, outs_t);
    }
}
